// SingleKPCN_15109694947679
// MI455X (gfx1250) — compile-verified, matrix operand order checked by reading, not run on hardware
//
#include <hip/hip_runtime.h>
#include <hip/hip_bf16.h>

// ---------------------------------------------------------------------------
// KPCN on MI455X (gfx1250): 9x (5x5 VALID conv) as shift-GEMM via
// v_wmma_f32_16x16x32_bf16 with TDM (tensor_load_to_lds) double-buffered
// weight staging, then per-pixel softmax(441) + 21x21 normalized filter.
// ---------------------------------------------------------------------------

typedef __attribute__((ext_vector_type(16))) __bf16 v16bf;
typedef __attribute__((ext_vector_type(8)))  float  v8f;
typedef __attribute__((ext_vector_type(4)))  unsigned int v4u;
typedef __attribute__((ext_vector_type(8)))  int   v8i;
typedef __attribute__((ext_vector_type(4)))  int   v4i;

union Frag16 { v16bf v; uint4 q[2]; };

// ---------------------------------------------------------------------------
// TDM: async 1-D copy of (elems8 * 8) bytes from global -> LDS.
// D# per CDNA5 ISA ch.8: group0 = {count/flags, lds_addr, global_addr, type},
// group1 = {data_size=8B, tensor_dim0=tile_dim0=elems8, dims1=1, stride=elems8}.
// Generic->LDS byte address: LDS aperture keeps the offset in addr[31:0].
// Toolchain is the 6-arg clang-23 form: (v4u, v8i, v4i, v4i, v8i, i32 cpol).
// ---------------------------------------------------------------------------
__device__ __forceinline__
void tdm_load_tile(const __bf16* gsrc, const __bf16* lds, int elems8)
{
    const unsigned long long ga = (unsigned long long)(uintptr_t)gsrc;
    v4u g0;
    g0.x = 1u;                                              // count=1, user D#
    g0.y = (unsigned int)(uintptr_t)lds;                    // lds_addr (bytes)
    g0.z = (unsigned int)ga;                                // global_addr[31:0]
    g0.w = (unsigned int)((ga >> 32) & 0x1FFFFFFull)        // global_addr[56:32]
         | (2u << 30);                                      // type = 2 (image)
    v8i g1;
    g1[0] = 0x3 << 16;                                      // data_size = 8 bytes
    g1[1] = elems8 << 16;                                   // tensor_dim0[15:0]
    g1[2] = 1 << 16;                                        // tensor_dim1 = 1
    g1[3] = elems8 << 16;                                   // tile_dim0
    g1[4] = 1;                                              // tile_dim1 = 1
    g1[5] = elems8;                                         // tensor_dim0_stride
    g1[6] = 0;
    g1[7] = 0;
    v4i z4 = {0, 0, 0, 0};                                  // <=2D: groups 2/3 zero
    v8i z8 = {0, 0, 0, 0, 0, 0, 0, 0};
    __builtin_amdgcn_tensor_load_to_lds(g0, g1, z4, z4, z8, 0);
}

// ---------------------------------------------------------------------------
// Convert kpcn_in (f32, [B=8][34][128][128]) -> bf16 [b][y][x][64] (zero-pad ch)
// ---------------------------------------------------------------------------
__global__ void cvt_input_kernel(const float* __restrict__ in, __bf16* __restrict__ out)
{
    int idx = blockIdx.x * blockDim.x + threadIdx.x;          // over 8*128*128*64
    if (idx >= 8 * 128 * 128 * 64) return;
    int c = idx & 63;
    int r = idx >> 6;
    int x = r & 127; r >>= 7;
    int y = r & 127; r >>= 7;
    int b = r;
    float v = 0.f;
    if (c < 34) v = in[(((size_t)b * 34 + c) * 128 + y) * 128 + x];
    out[idx] = (__bf16)v;
}

// ---------------------------------------------------------------------------
// Pack one conv layer's weights: f32 OIHW (cout,cin,5,5) ->
// bf16 [tap=ky*5+kx][m<coutP][k<cinP], zero-padded.
// ---------------------------------------------------------------------------
__global__ void pack_weights_kernel(const float* __restrict__ w, __bf16* __restrict__ wp,
                                    int cout, int cin, int coutP, int cinP)
{
    int idx = blockIdx.x * blockDim.x + threadIdx.x;          // over 25*coutP*cinP
    int total = 25 * coutP * cinP;
    if (idx >= total) return;
    int k = idx % cinP;
    int m = (idx / cinP) % coutP;
    int t = idx / (cinP * coutP);
    float v = 0.f;
    if (m < cout && k < cin) v = w[((size_t)m * cin + k) * 25 + t];
    wp[idx] = (__bf16)v;
}

// ---------------------------------------------------------------------------
// One conv layer as shift-GEMM with WMMA bf16 -> f32 accumulate.
// Block = 256 threads = 8 wave32. Each wave: 64 couts x 16 pixels
// (4 accumulators; one B fragment feeds 4 WMMAs -> 4x B-traffic reuse).
// Workgroup tile: 64 couts x 128 pixels.
// A tiles (64 x cinP weights per tap) are TDM-DMAed into double-buffered LDS
// one tap ahead of compute, synced via s_wait_tensorcnt + barriers.
// ---------------------------------------------------------------------------
template<bool RELU>
__global__ __launch_bounds__(256)
void conv_wmma_kernel(const __bf16* __restrict__ actIn,  // [B][Hi][Wi][cinP]
                      const __bf16* __restrict__ wpack,  // [25][coutP][cinP]
                      const float*  __restrict__ bias, int cout,
                      __bf16* __restrict__ actOut,       // [B][Ho][Wo][coutP]
                      int Hi, int Wi, int cinP, int Ho, int Wo, int coutP)
{
    __shared__ __bf16 smA[2 * 64 * 128];                 // double buffer, 32 KB

    const int tid    = threadIdx.x;
    const int lane   = tid & 31;
    const int wave   = tid >> 5;
    const int n      = lane & 15;                        // pixel col / A row (M)
    const int hiHalf = lane >> 4;                        // 0: K 0..7,16..23  1: K 8..15,24..31
    const int kofs   = hiHalf * 8;

    const int npix  = Ho * Wo;
    int pix = blockIdx.x * 128 + wave * 16 + n;
    const bool valid = (pix < npix);
    if (!valid) pix = npix - 1;                          // clamp (store is guarded)
    const int oy = pix / Wo;
    const int ox = pix - oy * Wo;

    const int mbase     = blockIdx.y * 64;               // 64-cout super-tile
    const int b         = blockIdx.z;
    const int ksteps    = cinP >> 5;
    const int tileElems = 64 * cinP;                     // bf16 elems per A tile
    const int elems8    = 16 * cinP;                     // 8-byte elems per A tile

    v8f acc0 = {}, acc1 = {}, acc2 = {}, acc3 = {};

    // Prologue: kick off tap 0 A-tile DMA.
    if (tid < 32)
        tdm_load_tile(wpack + ((size_t)0 * coutP + mbase) * cinP, smA, elems8);

    for (int t = 0; t < 25; ++t) {
        __syncthreads();                                 // done reading buf[(t+1)&1]
        if (tid < 32) {
            if (t < 24) {
                tdm_load_tile(wpack + ((size_t)(t + 1) * coutP + mbase) * cinP,
                              smA + ((t + 1) & 1) * tileElems, elems8);
                __builtin_amdgcn_s_wait_tensorcnt(1);    // tap t tile landed
            } else {
                __builtin_amdgcn_s_wait_tensorcnt(0);
            }
        }
        __syncthreads();                                 // LDS visible to all waves

        const __bf16* sm = smA + (t & 1) * tileElems;
        const int ky = t / 5, kx = t - ky * 5;
        const __bf16* bb =
            actIn + ((((size_t)b * Hi) + (oy + ky)) * Wi + (ox + kx)) * cinP;

        for (int ks = 0; ks < ksteps; ++ks) {
            Frag16 bf;
            const __bf16* bp = bb + ks * 32 + kofs;
            bf.q[0] = *(const uint4*)(bp);
            bf.q[1] = *(const uint4*)(bp + 16);

            Frag16 a;
            const __bf16* ap0 = sm + (size_t)n * cinP + ks * 32 + kofs;
            const size_t mstep = (size_t)16 * cinP;

            a.q[0] = *(const uint4*)(ap0);
            a.q[1] = *(const uint4*)(ap0 + 16);
            acc0 = __builtin_amdgcn_wmma_f32_16x16x32_bf16(
                       false, a.v, false, bf.v, (short)0, acc0, false, false);

            a.q[0] = *(const uint4*)(ap0 + mstep);
            a.q[1] = *(const uint4*)(ap0 + mstep + 16);
            acc1 = __builtin_amdgcn_wmma_f32_16x16x32_bf16(
                       false, a.v, false, bf.v, (short)0, acc1, false, false);

            a.q[0] = *(const uint4*)(ap0 + 2 * mstep);
            a.q[1] = *(const uint4*)(ap0 + 2 * mstep + 16);
            acc2 = __builtin_amdgcn_wmma_f32_16x16x32_bf16(
                       false, a.v, false, bf.v, (short)0, acc2, false, false);

            a.q[0] = *(const uint4*)(ap0 + 3 * mstep);
            a.q[1] = *(const uint4*)(ap0 + 3 * mstep + 16);
            acc3 = __builtin_amdgcn_wmma_f32_16x16x32_bf16(
                       false, a.v, false, bf.v, (short)0, acc3, false, false);
        }
    }

    if (valid) {
        const size_t ob =
            ((((size_t)b * Ho) + oy) * Wo + ox) * coutP + (size_t)mbase;
        v8f accs[4] = { acc0, acc1, acc2, acc3 };
#pragma unroll
        for (int ms = 0; ms < 4; ++ms) {
#pragma unroll
            for (int r = 0; r < 8; ++r) {
                const int m  = ms * 16 + r + hiHalf * 8; // D: vgpr r -> M=r / M=r+8
                const int gm = mbase + m;
                float v = accs[ms][r] + (gm < cout ? bias[gm] : 0.f);
                if (RELU) v = fmaxf(v, 0.f);
                actOut[ob + m] = (__bf16)v;
            }
        }
    }
}

// ---------------------------------------------------------------------------
// Per-pixel softmax over 441 kernel logits + 21x21 normalized filtering.
// kact: bf16 [b][y][x][448] (layer-8 output), buf: f32 (8,3,128,128),
// out: f32 (8,3,92,92).
// ---------------------------------------------------------------------------
__global__ void kpcn_filter_kernel(const __bf16* __restrict__ kact,
                                   const float*  __restrict__ buf,
                                   float* __restrict__ out)
{
    const int H = 92, W = 92, CROP = 18;
    int idx = blockIdx.x * blockDim.x + threadIdx.x;     // over 8*92*92
    if (idx >= 8 * H * W) return;
    int x = idx % W;
    int y = (idx / W) % H;
    int b = idx / (W * H);

    const __bf16* kp = kact + (size_t)idx * 448;

    float mx = -1e30f;
    for (int c = 0; c < 441; ++c) mx = fmaxf(mx, (float)kp[c]);
    float s = 0.f;
    for (int c = 0; c < 441; ++c) s += __expf((float)kp[c] - mx);
    const float inv = 1.f / s;

    float a0 = 0.f, a1 = 0.f, a2 = 0.f, wsum = 0.f;
    for (int i = 0; i < 21; ++i) {
        const int yy = y + i - 10;
        const bool vy = (yy >= 0) && (yy < H);
        for (int j = 0; j < 21; ++j) {
            const float w = __expf((float)kp[i * 21 + j] - mx) * inv;
            const int xx = x + j - 10;
            if (vy && xx >= 0 && xx < W) {
                wsum += w;
                const size_t base =
                    (((size_t)b * 3) * 128 + (CROP + yy)) * 128 + (CROP + xx);
                a0 += w * buf[base];
                a1 += w * buf[base + 128 * 128];
                a2 += w * buf[base + 2 * 128 * 128];
            }
        }
    }
    const float r = 1.f / wsum;
    const size_t o = (((size_t)b * 3) * H + y) * W + x;
    out[o]                     = a0 * r;
    out[o + (size_t)H * W]     = a1 * r;
    out[o + (size_t)2 * H * W] = a2 * r;
}

// ---------------------------------------------------------------------------
// Host-side orchestration.
// Workspace layout: [packed weights ~9 MB][actA 30 MB][actB 64 MB]
// ---------------------------------------------------------------------------
extern "C" void kernel_launch(void* const* d_in, const int* in_sizes, int n_in,
                              void* d_out, int out_size, void* d_ws, size_t ws_size,
                              hipStream_t stream)
{
    (void)in_sizes; (void)n_in; (void)out_size; (void)ws_size;

    const float* kpcn_in  = (const float*)d_in[0];
    const float* kpcn_buf = (const float*)d_in[1];

    // Per-layer geometry.
    int cin[9], cinP[9], cout_[9], coutP[9], Hi[9], Ho[9];
    size_t woffE[9];                                     // element offsets into wpack
    size_t wtotE = 0;
    for (int l = 0; l < 9; ++l) {
        cin[l]   = (l == 0) ? 34 : 100;
        cinP[l]  = (l == 0) ? 64 : 128;
        cout_[l] = (l == 8) ? 441 : 100;
        coutP[l] = (l == 8) ? 448 : 128;
        Hi[l]    = 128 - 4 * l;
        Ho[l]    = Hi[l] - 4;
        woffE[l] = wtotE;
        wtotE   += (size_t)25 * coutP[l] * cinP[l];
    }

    char*   ws    = (char*)d_ws;
    __bf16* wpack = (__bf16*)ws;
    size_t  aOff  = (wtotE * sizeof(__bf16) + 255) & ~(size_t)255;
    __bf16* actA  = (__bf16*)(ws + aOff);
    size_t  bOff  = aOff + (size_t)30 * 1024 * 1024;     // actA <= 29.5 MB
    __bf16* actB  = (__bf16*)(ws + bOff);                // actB <= 60.7 MB

    // 1) Pack all layer weights to bf16 tap-major tiles.
    for (int l = 0; l < 9; ++l) {
        int tot = 25 * coutP[l] * cinP[l];
        pack_weights_kernel<<<(tot + 255) / 256, 256, 0, stream>>>(
            (const float*)d_in[2 + l], wpack + woffE[l],
            cout_[l], cin[l], coutP[l], cinP[l]);
    }

    // 2) Convert input activations to bf16 channel-last (pad 34 -> 64).
    {
        int tot = 8 * 128 * 128 * 64;
        cvt_input_kernel<<<(tot + 255) / 256, 256, 0, stream>>>(kpcn_in, actA);
    }

    // 3) Conv chain, ping-pong A <-> B. Input sits in A, so even layers A->B.
    __bf16* bufs[2] = { actA, actB };
    for (int l = 0; l < 9; ++l) {
        const __bf16* src = bufs[l & 1];
        __bf16*       dst = bufs[(l + 1) & 1];
        const int npix = Ho[l] * Ho[l];
        dim3 grid((npix + 127) / 128, coutP[l] / 64, 8);
        if (l < 8) {
            conv_wmma_kernel<true><<<grid, 256, 0, stream>>>(
                src, wpack + woffE[l], (const float*)d_in[11 + l], cout_[l],
                dst, Hi[l], Hi[l], cinP[l], Ho[l], Ho[l], coutP[l]);
        } else {
            conv_wmma_kernel<false><<<grid, 256, 0, stream>>>(
                src, wpack + woffE[l], (const float*)d_in[11 + l], cout_[l],
                dst, Hi[l], Hi[l], cinP[l], Ho[l], Ho[l], coutP[l]);
        }
    }

    // 4) Softmax + 21x21 normalized kernel filtering. Layer-8 output is in actB.
    {
        int tot = 8 * 92 * 92;
        kpcn_filter_kernel<<<(tot + 255) / 256, 256, 0, stream>>>(
            actB, kpcn_buf, (float*)d_out);
    }
}
